// MultiHeadAttention_11484742550065
// MI455X (gfx1250) — matrix operand order repaired and checked by reading, not compiled, not run on hardware
//
#include <hip/hip_runtime.h>
#include <hip/hip_bf16.h>

// ---------------------------------------------------------------------------
// MultiHeadAttention forward for MI455X (gfx1250), wave32 + WMMA f16->f32.
//   B=8, T=2048, C=1024, H=16, d=64
// Pipeline:
//   k0: x (f32) -> xh (f16)
//   k1: W_qkv (C,3C f32) -> Wt (3C,C f16, transposed)
//   k2: qkv GEMM with v_wmma_f32_16x16x32_f16; epilogue adds bias and scatters
//       q*d^-0.5 -> qh (B,H,T,d) f16 ; k -> kh (B,H,T,d) f16 ; v -> vt (B,H,d,T) f16
//   k3: flash attention per (b,h,64 q rows): online softmax, all GEMMs on WMMA
// ---------------------------------------------------------------------------

#define B_  8
#define T_  2048
#define C_  1024
#define H_  16
#define D_  64
#define NEG_INF (-__builtin_inff())

typedef __attribute__((ext_vector_type(16))) _Float16 v16h;  // 8 VGPRs
typedef __attribute__((ext_vector_type(8)))  float    v8f;   // 8 VGPRs
typedef __attribute__((ext_vector_type(4)))  _Float16 h4;

// ---- WMMA wrapper ----------------------------------------------------------
__device__ __forceinline__ v8f wmma_f16(v16h a, v16h b, v8f c) {
  // D = A(16x32 f16) * B(32x16 f16) + C(16x16 f32)
  return __builtin_amdgcn_wmma_f32_16x16x32_f16(
      /*neg_a=*/false, a, /*neg_b=*/false, b,
      /*c_mod=*/(short)0, c, /*reuse_a=*/false, /*reuse_b=*/false);
}

// ---- Fragment builders from LDS tiles --------------------------------------
// A (16x32, f16), row-major tile in LDS (stride in halves, 16B-aligned rows).
// ISA layout: lanes 0-15: M=lane, halves = K[cb+0..7], K[cb+16..23]
//             lanes16-31: M=lane-15? no: M=lane&15, K[cb+8..15], K[cb+24..31]
__device__ __forceinline__ v16h frag_a(const _Float16* lds, int row0, int stride, int cb) {
  const int lane = threadIdx.x & 31;
  const int r  = row0 + (lane & 15);
  const int kb = cb + ((lane >> 4) << 3);        // 0 or 8 (+cb)
  const _Float16* p = lds + r * stride + kb;
  union { v16h v; uint4 u[2]; } t;
  t.u[0] = *(const uint4*)(p);
  t.u[1] = *(const uint4*)(p + 16);
  return t.v;
}
// B (32x16, f16) built from a transposed tile Bt[n][k] in LDS.
// lanes 0-15: N=lane, halves = K[cb+0..7], K[cb+16..23]; lanes 16-31: N=lane&15, K[cb+8..15], K[cb+24..31]
__device__ __forceinline__ v16h frag_b(const _Float16* lds, int n0, int stride, int cb) {
  const int lane = threadIdx.x & 31;
  const int n  = n0 + (lane & 15);
  const int kb = cb + ((lane >> 4) << 3);        // 0 or 8 (+cb)
  const _Float16* p = lds + n * stride + kb;
  union { v16h v; uint4 u[2]; } t;
  t.u[0] = *(const uint4*)(p);
  t.u[1] = *(const uint4*)(p + 16);
  return t.v;
}

// ---- k0: f32 -> f16 convert (vector x4) ------------------------------------
__global__ __launch_bounds__(256) void cvt_x_kernel(const float* __restrict__ x,
                                                    _Float16* __restrict__ xh, int n4) {
  int i = blockIdx.x * blockDim.x + threadIdx.x;
  if (i < n4) {
    float4 v = ((const float4*)x)[i];
    h4 o = {(_Float16)v.x, (_Float16)v.y, (_Float16)v.z, (_Float16)v.w};
    ((h4*)xh)[i] = o;
  }
}

// ---- k1: W (C,3C) f32 -> Wt (3C,C) f16 -------------------------------------
__global__ __launch_bounds__(256) void cvt_wt_kernel(const float* __restrict__ w,
                                                     _Float16* __restrict__ wt) {
  int i = blockIdx.x * blockDim.x + threadIdx.x;   // over 3C*C
  int n = i >> 10;          // 0..3071
  int k = i & 1023;         // 0..1023
  wt[i] = (_Float16)w[k * (3 * C_) + n];
}

// ---- k2: QKV GEMM: (16384 x 1024) * (1024 x 3072) --------------------------
#define GM 128
#define GN 64
#define GK 32
#define ASTR 40   // halves; 80B rows, 16B aligned
#define BSTR 40

__global__ __launch_bounds__(256) void qkv_gemm_kernel(
    const _Float16* __restrict__ xh, const _Float16* __restrict__ wt,
    const float* __restrict__ bias,
    _Float16* __restrict__ qh, _Float16* __restrict__ kh, _Float16* __restrict__ vt) {
  __shared__ __align__(16) _Float16 As[GM * ASTR];
  __shared__ __align__(16) _Float16 Bs[GN * BSTR];

  const int tid  = threadIdx.x;
  const int lane = tid & 31;
  const int wid  = tid >> 5;          // 0..7
  const int wm   = wid >> 1;          // 0..3 -> M offset wm*32
  const int wn   = wid & 1;           // 0..1 -> N offset wn*32
  const int m0   = blockIdx.y * GM;
  const int n0   = blockIdx.x * GN;

  v8f acc[2][2] = {};

  for (int k0 = 0; k0 < C_; k0 += GK) {
    __syncthreads();
#pragma unroll
    for (int i = 0; i < 2; ++i) {     // A tile: 128x32 halves = 512 x 16B chunks
      int c = tid + i * 256;
      int r = c >> 2, col = (c & 3) << 3;
      *(uint4*)&As[r * ASTR + col] = *(const uint4*)&xh[(size_t)(m0 + r) * C_ + k0 + col];
    }
    {                                 // B tile: 64x32 halves = 256 chunks
      int r = tid >> 2, col = (tid & 3) << 3;
      *(uint4*)&Bs[r * BSTR + col] = *(const uint4*)&wt[(size_t)(n0 + r) * C_ + k0 + col];
    }
    if (k0 + GK < C_) {               // speculative prefetch of next tiles
      __builtin_prefetch(&xh[(size_t)(m0 + (tid >> 1)) * C_ + k0 + GK], 0, 1);
      __builtin_prefetch(&wt[(size_t)(n0 + (tid >> 2)) * C_ + k0 + GK], 0, 1);
    }
    __syncthreads();

    v16h a0 = frag_a(As, wm * 32,      ASTR, 0);
    v16h a1 = frag_a(As, wm * 32 + 16, ASTR, 0);
    v16h b0 = frag_b(Bs, wn * 32,      BSTR, 0);
    v16h b1 = frag_b(Bs, wn * 32 + 16, BSTR, 0);
    acc[0][0] = wmma_f16(a0, b0, acc[0][0]);
    acc[0][1] = wmma_f16(a0, b1, acc[0][1]);
    acc[1][0] = wmma_f16(a1, b0, acc[1][0]);
    acc[1][1] = wmma_f16(a1, b1, acc[1][1]);
  }

  // Epilogue: bias add + scatter into q / k / v^T (region uniform per block).
  const int halfsel = (lane >> 4) << 3;   // C-layout: lanes>=16 hold rows M+8
  const int nl = lane & 15;
#pragma unroll
  for (int mi = 0; mi < 2; ++mi) {
#pragma unroll
    for (int ni = 0; ni < 2; ++ni) {
      int N = n0 + wn * 32 + ni * 16 + nl;   // global col in [0,3072)
      float bv = bias[N];
      int region = N >> 10;                  // 0=q 1=k 2=v
      int cc = N & 1023, h = cc >> 6, dd = cc & 63;
#pragma unroll
      for (int j = 0; j < 8; ++j) {
        int M = m0 + wm * 32 + mi * 16 + j + halfsel;
        int b = M >> 11, t = M & 2047;
        float v = acc[mi][ni][j] + bv;
        size_t bh = (size_t)(b * H_ + h);
        if (region == 0) {
          qh[(bh * T_ + t) * D_ + dd] = (_Float16)(v * 0.125f); // fold d^-0.5
        } else if (region == 1) {
          kh[(bh * T_ + t) * D_ + dd] = (_Float16)v;
        } else {
          vt[(bh * D_ + dd) * T_ + t] = (_Float16)v;            // transposed V
        }
      }
    }
  }
}

// ---- k3: flash attention ----------------------------------------------------
#define QSTR 72   // 144B rows, 16B aligned, bank-friendly

__global__ __launch_bounds__(128) void attn_kernel(
    const _Float16* __restrict__ qh, const _Float16* __restrict__ kh,
    const _Float16* __restrict__ vt, const int* __restrict__ mask,
    float* __restrict__ out) {
  __shared__ __align__(16) _Float16 Qs[64 * QSTR];
  __shared__ __align__(16) _Float16 Ks[64 * QSTR];
  __shared__ __align__(16) _Float16 Vs[64 * QSTR];
  __shared__ __align__(16) _Float16 Ps[4][16 * QSTR];  // per-wave P buffer

  const int tid  = threadIdx.x;
  const int lane = tid & 31;
  const int w    = tid >> 5;                  // wave 0..3 -> q rows w*16..+15
  const int nl   = lane & 15;
  const int rowoff = (lane >> 4) << 3;

  const int bid  = blockIdx.x;
  const int qblk = bid & 31;
  const int bh   = bid >> 5;                  // b*16+h
  const int b    = bh >> 4;
  const int q0   = qblk * 64;

  const _Float16* qbase = qh + (size_t)bh * T_ * D_;
  const _Float16* kbase = kh + (size_t)bh * T_ * D_;
  const _Float16* vbase = vt + (size_t)bh * D_ * T_;
  const int* mbase = mask + b * T_;

  // Load Q tile (64x64) once.
#pragma unroll
  for (int i = 0; i < 4; ++i) {
    int c = tid + i * 128;
    int r = c >> 3, col = (c & 7) << 3;
    *(uint4*)&Qs[r * QSTR + col] = *(const uint4*)&qbase[(size_t)(q0 + r) * D_ + col];
  }

  v8f o[4] = {};
  float m[8], l[8];
#pragma unroll
  for (int j = 0; j < 8; ++j) { m[j] = NEG_INF; l[j] = 0.f; }

  for (int kb = 0; kb < T_ / 64; ++kb) {
    const int key0 = kb * 64;
    __syncthreads();
#pragma unroll
    for (int i = 0; i < 4; ++i) {            // K tile 64x64, Vt tile 64x64
      int c = tid + i * 128;
      int r = c >> 3, col = (c & 7) << 3;
      *(uint4*)&Ks[r * QSTR + col] = *(const uint4*)&kbase[(size_t)(key0 + r) * D_ + col];
      *(uint4*)&Vs[r * QSTR + col] = *(const uint4*)&vbase[(size_t)r * T_ + key0 + col];
    }
    int mv[4];
#pragma unroll
    for (int nt = 0; nt < 4; ++nt) mv[nt] = mbase[key0 + nt * 16 + nl];
    __syncthreads();

    // S = Q * K^T  (K-dim = d = 64 -> two k-steps of 32)
    v16h aq0 = frag_a(Qs, w * 16, QSTR, 0);
    v16h aq1 = frag_a(Qs, w * 16, QSTR, 32);
    v8f s[4];
#pragma unroll
    for (int nt = 0; nt < 4; ++nt) {
      v8f z = {};
      z = wmma_f16(aq0, frag_b(Ks, nt * 16, QSTR, 0),  z);
      z = wmma_f16(aq1, frag_b(Ks, nt * 16, QSTR, 32), z);
      s[nt] = z;
    }
#pragma unroll
    for (int nt = 0; nt < 4; ++nt)
#pragma unroll
      for (int j = 0; j < 8; ++j)
        s[nt][j] = mv[nt] ? s[nt][j] : NEG_INF;

    // Online softmax: row = (j + rowoff); one row spans 16 lanes of a half.
    float mnew[8], sc[8];
#pragma unroll
    for (int j = 0; j < 8; ++j) {
      float r = fmaxf(fmaxf(s[0][j], s[1][j]), fmaxf(s[2][j], s[3][j]));
      r = fmaxf(r, __shfl_xor(r, 1, 32));
      r = fmaxf(r, __shfl_xor(r, 2, 32));
      r = fmaxf(r, __shfl_xor(r, 4, 32));
      r = fmaxf(r, __shfl_xor(r, 8, 32));
      float mn = fmaxf(m[j], r);
      mnew[j] = mn;
      sc[j] = (m[j] == mn) ? 1.f : __expf(m[j] - mn);
    }
#pragma unroll
    for (int j = 0; j < 8; ++j) {
      float mn = mnew[j];
      float ps = 0.f;
#pragma unroll
      for (int nt = 0; nt < 4; ++nt) {
        float pv = (mn == NEG_INF) ? 0.f : __expf(s[nt][j] - mn);
        ps += pv;
        Ps[w][(j + rowoff) * QSTR + nt * 16 + nl] = (_Float16)pv; // C->A relayout
      }
      ps += __shfl_xor(ps, 1, 32);
      ps += __shfl_xor(ps, 2, 32);
      ps += __shfl_xor(ps, 4, 32);
      ps += __shfl_xor(ps, 8, 32);
      l[j] = l[j] * sc[j] + ps;
      m[j] = mn;
    }
#pragma unroll
    for (int nt = 0; nt < 4; ++nt)
#pragma unroll
      for (int j = 0; j < 8; ++j) o[nt][j] *= sc[j];

    // O += P * V   (LDS is in-order per wave: writes above visible to reads)
    v16h ap0 = frag_a(Ps[w], 0, QSTR, 0);
    v16h ap1 = frag_a(Ps[w], 0, QSTR, 32);
#pragma unroll
    for (int nt = 0; nt < 4; ++nt) {
      o[nt] = wmma_f16(ap0, frag_b(Vs, nt * 16, QSTR, 0),  o[nt]);
      o[nt] = wmma_f16(ap1, frag_b(Vs, nt * 16, QSTR, 32), o[nt]);
    }
  }

  // Store O / l  -> flat (B,H,T,d) f32 (matches reference's reshape semantics)
  float* obase = out + ((size_t)bh * T_ + q0 + w * 16) * D_;
#pragma unroll
  for (int nt = 0; nt < 4; ++nt) {
    int dd = nt * 16 + nl;
#pragma unroll
    for (int j = 0; j < 8; ++j)
      obase[(size_t)(j + rowoff) * D_ + dd] = o[nt][j] / l[j];
  }
}

// ---------------------------------------------------------------------------
extern "C" void kernel_launch(void* const* d_in, const int* in_sizes, int n_in,
                              void* d_out, int out_size, void* d_ws, size_t ws_size,
                              hipStream_t stream) {
  const float* x    = (const float*)d_in[0];   // (B,T,C)
  const int*   mask = (const int*)d_in[1];     // (B,1,1,T)
  const float* wq   = (const float*)d_in[2];   // (C,3C)
  const float* bq   = (const float*)d_in[3];   // (3C)
  float* out = (float*)d_out;

  // Workspace layout (bytes, 256-aligned)
  const size_t XH_OFF = 0;                                  // 16384*1024 f16
  const size_t WT_OFF = XH_OFF + (size_t)B_ * T_ * C_ * 2;  // 3072*1024 f16
  const size_t QH_OFF = WT_OFF + (size_t)3 * C_ * C_ * 2;   // (B,H,T,d) f16
  const size_t KH_OFF = QH_OFF + (size_t)B_ * T_ * C_ * 2;
  const size_t VT_OFF = KH_OFF + (size_t)B_ * T_ * C_ * 2;
  const size_t NEED   = VT_OFF + (size_t)B_ * T_ * C_ * 2;  // ~140.5 MB
  if (ws_size < NEED) return;

  char* ws = (char*)d_ws;
  _Float16* xh = (_Float16*)(ws + XH_OFF);
  _Float16* wt = (_Float16*)(ws + WT_OFF);
  _Float16* qh = (_Float16*)(ws + QH_OFF);
  _Float16* kh = (_Float16*)(ws + KH_OFF);
  _Float16* vt = (_Float16*)(ws + VT_OFF);

  // k0: x -> f16 (vectorized by 4)
  {
    int n4 = (B_ * T_ * C_) / 4;
    cvt_x_kernel<<<(n4 + 255) / 256, 256, 0, stream>>>(x, xh, n4);
  }
  // k1: W -> Wt f16
  {
    int n = 3 * C_ * C_;
    cvt_wt_kernel<<<(n + 255) / 256, 256, 0, stream>>>(wq, wt);
  }
  // k2: QKV GEMM + scatter
  {
    dim3 grid((3 * C_) / GN, (B_ * T_) / GM);   // (48, 128)
    qkv_gemm_kernel<<<grid, 256, 0, stream>>>(xh, wt, bq, qh, kh, vt);
  }
  // k3: flash attention
  {
    int grid = B_ * H_ * (T_ / 64);             // 4096
    attn_kernel<<<grid, 128, 0, stream>>>(qh, kh, vt, mask, out);
  }
}
